// MushroomBodyNetwork_14439680049866
// MI455X (gfx1250) — matrix operand order repaired and checked by reading, not compiled, not run on hardware
//
#include <hip/hip_runtime.h>
#include <hip/hip_bf16.h>

typedef __attribute__((ext_vector_type(16))) _Float16 v16h;
typedef __attribute__((ext_vector_type(8)))  _Float16 h8;
typedef __attribute__((ext_vector_type(8)))  float    v8f;
typedef __attribute__((ext_vector_type(4)))  float    f4;

#define N_VPN   2048
#define N_KC    32768
#define BATCH   2048
#define NUM_ACT 1638

#define TILE_M 128
#define TILE_N 128
#define TILE_K 32
#define LDS_K  40   // padded half-stride: 80 bytes, multiple of 16B

// ---------------------------------------------------------------------------
// Kernel A: scores = x @ W_mb^T  via v_wmma_f32_16x16x32_f16
// grid = (M/128, N/128), block = 256 threads (8 waves; 4x2 wave grid),
// each wave owns a 32x64 patch = 2x4 WMMA 16x16 tiles.
// Register-prefetch + double-buffered LDS: one barrier per K-step, global
// loads for step k+1 are in flight while step k's WMMAs execute.
// ---------------------------------------------------------------------------
__global__ __launch_bounds__(256) void mb_gemm_kernel(
    const float* __restrict__ x, const float* __restrict__ W,
    float* __restrict__ scores) {
  __shared__ __align__(16) _Float16 As[2][TILE_M][LDS_K];
  __shared__ __align__(16) _Float16 Bs[2][TILE_N][LDS_K];

  const int m0   = blockIdx.x * TILE_M;   // fast axis = M: W band read once
  const int n0   = blockIdx.y * TILE_N;
  const int t    = threadIdx.x;
  const int lane = t & 31;
  const int wave = t >> 5;
  const int wm   = wave >> 1;             // 0..3 -> 32-row strip
  const int wn   = wave & 1;              // 0..1 -> 64-col strip

  v8f acc[2][4];
#pragma unroll
  for (int i = 0; i < 2; ++i)
#pragma unroll
    for (int j = 0; j < 4; ++j) acc[i][j] = (v8f)(0.0f);

  // staging: 2 threads per row, 16 floats each
  const int srow = t >> 1;                // 0..127
  const int scol = (t & 1) * 16;          // 0 or 16

  const int ln15 = lane & 15;
  const int kbA  = (lane >> 4) * 8;       // A: halves {kb..kb+7, kb+16..kb+23}
  const int kbB  = (lane >> 4) * 8;       // B: halves {kb..kb+7, kb+16..kb+23}

  const float* ga = x + (size_t)(m0 + srow) * N_VPN + scol;
  const float* gb = W + (size_t)(n0 + srow) * N_VPN + scol;

  // prime the register prefetch with K-slice 0
  f4 av[4], bv[4];
#pragma unroll
  for (int j = 0; j < 4; ++j) {
    av[j] = ((const f4*)ga)[j];
    bv[j] = ((const f4*)gb)[j];
  }

  int buf = 0;
  for (int kk = 0; kk < N_VPN; kk += TILE_K, buf ^= 1) {
    // ---- commit prefetched K-slice to LDS[buf], f32 -> f16 ----
#pragma unroll
    for (int j = 0; j < 4; ++j)
#pragma unroll
      for (int e = 0; e < 4; ++e) {
        As[buf][srow][scol + j * 4 + e] = (_Float16)av[j][e];
        Bs[buf][srow][scol + j * 4 + e] = (_Float16)bv[j][e];
      }
    __syncthreads();   // single barrier per K-step (double-buffered LDS)

    // ---- kick off global loads of the next K-slice (hidden under WMMA) ----
    if (kk + TILE_K < N_VPN) {
      const float* gan = ga + kk + TILE_K;
      const float* gbn = gb + kk + TILE_K;
#pragma unroll
      for (int j = 0; j < 4; ++j) {
        av[j] = ((const f4*)gan)[j];
        bv[j] = ((const f4*)gbn)[j];
      }
    }

    // ---- build A fragments (ISA 16-bit A 16x32 layout) ----
    v16h afrag[2];
#pragma unroll
    for (int im = 0; im < 2; ++im) {
      const int m = wm * 32 + im * 16 + ln15;
      h8 lo = *(const h8*)&As[buf][m][kbA];
      h8 hi = *(const h8*)&As[buf][m][kbA + 16];
#pragma unroll
      for (int e = 0; e < 8; ++e) { afrag[im][e] = lo[e]; afrag[im][8 + e] = hi[e]; }
    }

    // ---- B fragments + 8 WMMAs ----
#pragma unroll
    for (int jn = 0; jn < 4; ++jn) {
      const int n = wn * 64 + jn * 16 + ln15;
      h8 blo = *(const h8*)&Bs[buf][n][kbB];
      h8 bhi = *(const h8*)&Bs[buf][n][kbB + 16];
      v16h bfrag;
#pragma unroll
      for (int e = 0; e < 8; ++e) { bfrag[e] = blo[e]; bfrag[8 + e] = bhi[e]; }
#pragma unroll
      for (int im = 0; im < 2; ++im) {
        acc[im][jn] = __builtin_amdgcn_wmma_f32_16x16x32_f16(
            /*neg_a=*/false, afrag[im], /*neg_b=*/false, bfrag,
            /*c_mod=*/(short)0, acc[im][jn], /*reuse_a=*/false, /*reuse_b=*/false);
      }
    }
    // no trailing barrier: next iteration writes the *other* LDS buffer, and
    // writes to this buffer (step kk+2) occur only after the next barrier.
  }

  // ---- write C (ISA 32-bit C/D 16x16 layout: VGPR r -> M=r+8*(lane>=16)) ----
#pragma unroll
  for (int im = 0; im < 2; ++im) {
#pragma unroll
    for (int jn = 0; jn < 4; ++jn) {
      const int mbase = m0 + wm * 32 + im * 16 + (lane >> 4) * 8;
      const int col   = n0 + wn * 64 + jn * 16 + ln15;
#pragma unroll
      for (int r = 0; r < 8; ++r)
        scores[(size_t)(mbase + r) * N_KC + col] = acc[im][jn][r];
    }
  }
}

// ---------------------------------------------------------------------------
// Kernel B: per-row top-NUM_ACT binary mask via 4-pass LDS radix select,
// stable tie-break by lowest index (block prefix scan), in-place over scores.
// One block (1024 threads) per batch row; row keys live in 128 KB dynamic LDS
// (WGP has 320 KB on CDNA5).
// ---------------------------------------------------------------------------
__global__ __launch_bounds__(1024) void mb_select_kernel(
    float* __restrict__ out, const float* __restrict__ w_mbon) {
  extern __shared__ unsigned int keys[];            // N_KC u32 keys (128 KB)
  __shared__ unsigned int hist[256];
  __shared__ unsigned int scanbuf[1024];
  __shared__ unsigned int s_pref, s_need;

  const int b = blockIdx.x;
  const int t = threadIdx.x;
  float* row = out + (size_t)b * N_KC;

  // load scores, map to order-preserving uint keys (coalesced, conflict-free)
#pragma unroll 4
  for (int i = 0; i < N_KC / 1024; ++i) {
    const int idx = t + i * 1024;
    unsigned u = __float_as_uint(row[idx]);
    u ^= (u & 0x80000000u) ? 0xFFFFFFFFu : 0x80000000u;
    keys[idx] = u;
  }
  if (t == 0) { s_pref = 0u; s_need = NUM_ACT; }
  __syncthreads();

  // MSB-first radix select for the NUM_ACT-th largest key
  for (int pass = 0; pass < 4; ++pass) {
    if (t < 256) hist[t] = 0u;
    __syncthreads();
    const unsigned shift = 24u - 8u * pass;
    const unsigned pmask = (pass == 0) ? 0u : (0xFFFFFFFFu << (shift + 8u));
    const unsigned pcur  = s_pref;
    for (int i = 0; i < N_KC / 1024; ++i) {
      const unsigned u = keys[t + i * 1024];      // strided: bank-conflict-free
      if ((u & pmask) == pcur)
        atomicAdd(&hist[(u >> shift) & 0xFFu], 1u);
    }
    __syncthreads();
    if (t == 0) {
      unsigned need = s_need, accum = 0u;
      for (int bb = 255; bb >= 0; --bb) {
        const unsigned c = hist[bb];
        if (accum + c >= need) {
          s_need = need - accum;
          s_pref = pcur | ((unsigned)bb << shift);
          break;
        }
        accum += c;
      }
    }
    __syncthreads();
  }
  const unsigned T = s_pref;   // threshold key
  const unsigned r = s_need;   // how many keys == T to keep (lowest index first)

  // stable tie-break: exclusive scan of per-thread equal-counts (contiguous map)
  unsigned cnt = 0;
  for (int i = 0; i < N_KC / 1024; ++i)
    if (keys[t * (N_KC / 1024) + i] == T) ++cnt;
  scanbuf[t] = cnt;
  __syncthreads();
  for (int off = 1; off < 1024; off <<= 1) {
    const unsigned v = (t >= off) ? scanbuf[t - off] : 0u;
    __syncthreads();
    scanbuf[t] += v;
    __syncthreads();
  }
  unsigned rank = scanbuf[t] - cnt;   // exclusive prefix = index-ordered rank

  // write binary mask in place
  for (int i = 0; i < N_KC / 1024; ++i) {
    const int idx = t * (N_KC / 1024) + i;
    const unsigned u = keys[idx];
    float m = 0.0f;
    if (u > T) m = 1.0f;
    else if (u == T) { if (rank < r) m = 1.0f; ++rank; }
    row[idx] = m;
  }

  // mbon readout: NUM_ACT selected units, uniform weight
  if (t == 0)
    out[(size_t)BATCH * N_KC + b] = (float)NUM_ACT * w_mbon[0];
}

// ---------------------------------------------------------------------------
extern "C" void kernel_launch(void* const* d_in, const int* in_sizes, int n_in,
                              void* d_out, int out_size, void* d_ws, size_t ws_size,
                              hipStream_t stream) {
  const float* x      = (const float*)d_in[0];   // [BATCH, N_VPN]
  const float* W_mb   = (const float*)d_in[1];   // [N_KC, N_VPN]
  const float* w_mbon = (const float*)d_in[2];   // [1, N_KC]
  float* out = (float*)d_out;                    // mask [BATCH,N_KC] ++ mbon [BATCH,1]

  (void)in_sizes; (void)n_in; (void)d_ws; (void)ws_size; (void)out_size;

  // Phase 1: WMMA GEMM — scores parked in the mask region of d_out.
  dim3 grid_g(BATCH / TILE_M, N_KC / TILE_N);    // M fast => W_mb streamed once
  mb_gemm_kernel<<<grid_g, 256, 0, stream>>>(x, W_mb, out);

  // Phase 2: per-row radix top-k -> binary mask (in place) + mbon readout.
  mb_select_kernel<<<BATCH, 1024, N_KC * sizeof(unsigned int), stream>>>(out, w_mbon);
}
